// MolGPS_42709154791945
// MI455X (gfx1250) — matrix operand order repaired and checked by reading, not compiled, not run on hardware
//
#include <hip/hip_runtime.h>

// ---- problem constants (match reference) ----
#define C_    256
#define GN_   65536
#define E_    524288
#define L_    4
#define G_    512
#define N_    128
#define H_    4
#define DH_   64
#define OUT_  768

typedef __attribute__((ext_vector_type(16))) _Float16 v16h;
typedef __attribute__((ext_vector_type(8)))  float    v8f;

union Frag {
  v16h  h;
  uint4 q[2];
};

__device__ inline v8f wmma_f16(const Frag& a, const Frag& b, v8f c) {
  // v_wmma_f32_16x16x32_f16: D = A(16x32) * B(32x16) + C, f32 accumulate
  return __builtin_amdgcn_wmma_f32_16x16x32_f16(false, a.h, false, b.h,
                                                (short)0, c, false, false);
}

__device__ inline void atomic_add_f32(float* p, float v) {
  __hip_atomic_fetch_add(p, v, __ATOMIC_RELAXED, __HIP_MEMORY_SCOPE_AGENT);
}

// ---- CDNA5 async global->LDS DMA (ASYNCcnt-tracked) ----
__device__ inline void async_copy_b128(const void* g, void* lds) {
  // VDST = 32-bit LDS byte address, VADDR = 64-bit global address
  asm volatile("global_load_async_to_lds_b128 %0, %1, off"
               :
               : "v"((unsigned)(unsigned long long)lds),
                 "v"((unsigned long long)g)
               : "memory");
}

__device__ inline void wait_async0() {
#if __has_builtin(__builtin_amdgcn_s_wait_asynccnt)
  __builtin_amdgcn_s_wait_asynccnt(0);
#else
  asm volatile("s_wait_asynccnt 0x0" ::: "memory");
#endif
}

// ---------------------------------------------------------------------------
// Generic WMMA GEMM: out[M,N] = A[M,K] (f16) * W[N,K]^T (f16) + bias
// block tile 128x64, 8 waves (4x2), wave tile 32x32 (2x2 wmma accumulators)
// A/W tiles double-buffered in LDS, filled by async global->LDS B128 DMA.
// ---------------------------------------------------------------------------
#define TM 128
#define TN 64
#define LDT 48   // LDS tile stride in halves (96B rows: every 16B chunk aligned)

__device__ inline void stage_tileA(const _Float16* __restrict__ A, _Float16* As,
                                   long rowBase, int K, int kb, int t) {
  // 128 rows x 32 halves = 512 x 16B chunks -> 2 per thread
  #pragma unroll
  for (int i = 0; i < 2; ++i) {
    int idx = t + i * 256;
    int row = idx >> 2, c4 = idx & 3;
    async_copy_b128(A + (rowBase + row) * (long)K + kb + c4 * 8,
                    As + row * LDT + c4 * 8);
  }
}

__device__ inline void stage_tileW(const _Float16* __restrict__ W, _Float16* Ws,
                                   long colBase, int K, int kb, int t) {
  // 64 rows x 32 halves = 256 x 16B chunks -> 1 per thread
  int row = t >> 2, c4 = t & 3;
  async_copy_b128(W + (colBase + row) * (long)K + kb + c4 * 8,
                  Ws + row * LDT + c4 * 8);
}

__global__ __launch_bounds__(256) void gemm_f16(
    const _Float16* __restrict__ A, const _Float16* __restrict__ W,
    const float* __restrict__ bias,
    float* __restrict__ outF, _Float16* __restrict__ outH,
    const float* __restrict__ res,
    int M, int N, int K, int relu)
{
  __shared__ _Float16 As[2][TM * LDT];
  __shared__ _Float16 Ws[2][TN * LDT];

  const int t    = threadIdx.x;
  const int lane = t & 31;
  const int w    = t >> 5;
  const int wm   = w & 3;      // 4 wave rows
  const int wn   = w >> 2;     // 2 wave cols
  const int half = lane >> 4;
  const int l16  = lane & 15;
  const long rowBase = (long)blockIdx.x * TM;
  const long colBase = (long)blockIdx.y * TN;

  v8f acc[2][2] = {};

  // prologue: DMA first K-tile
  stage_tileA(A, As[0], rowBase, K, 0, t);
  stage_tileW(W, Ws[0], colBase, K, 0, t);
  wait_async0();
  __syncthreads();

  int cur = 0;
  for (int kb = 0; kb < K; kb += 32) {
    int nxt = cur ^ 1;
    if (kb + 32 < K) {           // prefetch next tile while computing
      stage_tileA(A, As[nxt], rowBase, K, kb + 32, t);
      stage_tileW(W, Ws[nxt], colBase, K, kb + 32, t);
    }

    Frag a[2], b[2];
    #pragma unroll
    for (int i = 0; i < 2; ++i) {
      int row = wm * 32 + i * 16 + l16;         // A rows: M = lane%16
      a[i].q[0] = *(const uint4*)(As[cur] + row * LDT + half * 8);
      a[i].q[1] = *(const uint4*)(As[cur] + row * LDT + 16 + half * 8);
    }
    #pragma unroll
    for (int j = 0; j < 2; ++j) {
      int n = wn * 32 + j * 16 + l16;           // B cols: N = lane%16
      b[j].q[0] = *(const uint4*)(Ws[cur] + n * LDT + half * 8);
      b[j].q[1] = *(const uint4*)(Ws[cur] + n * LDT + 16 + half * 8);
    }
    #pragma unroll
    for (int i = 0; i < 2; ++i)
      #pragma unroll
      for (int j = 0; j < 2; ++j)
        acc[i][j] = wmma_f16(a[i], b[j], acc[i][j]);

    wait_async0();
    __syncthreads();
    cur = nxt;
  }

  // epilogue: C/D layout -> VGPR r: M=r (lanes 0-15), M=r+8 (lanes 16-31)
  #pragma unroll
  for (int i = 0; i < 2; ++i) {
    #pragma unroll
    for (int r = 0; r < 8; ++r) {
      long m = rowBase + wm * 32 + i * 16 + r + half * 8;
      #pragma unroll
      for (int j = 0; j < 2; ++j) {
        int n = (int)colBase + wn * 32 + j * 16 + l16;
        float v = acc[i][j][r];
        if (bias) v += bias[n];
        if (res)  v += res[m * N + n];
        if (relu) v = fmaxf(v, 0.0f);
        if (outF) outF[m * N + n] = v;
        if (outH) outH[m * N + n] = (_Float16)v;
      }
    }
  }
}

// ---------------------------------------------------------------------------
// GINE edge GEMM: msg = relu(e @ W^T + bias + h[src]); atomic scatter-add
// into agg[dst]. M = E, N = K = C.
// ---------------------------------------------------------------------------
__global__ __launch_bounds__(256) void gemm_gine(
    const _Float16* __restrict__ A, const _Float16* __restrict__ W,
    const float* __restrict__ bias,
    const float* __restrict__ h,
    const int* __restrict__ src, const int* __restrict__ dst,
    float* __restrict__ agg, int M, int K)
{
  __shared__ _Float16 As[2][TM * LDT];
  __shared__ _Float16 Ws[2][TN * LDT];

  const int t    = threadIdx.x;
  const int lane = t & 31;
  const int w    = t >> 5;
  const int wm   = w & 3;
  const int wn   = w >> 2;
  const int half = lane >> 4;
  const int l16  = lane & 15;
  const long rowBase = (long)blockIdx.x * TM;
  const long colBase = (long)blockIdx.y * TN;

  v8f acc[2][2] = {};

  stage_tileA(A, As[0], rowBase, K, 0, t);
  stage_tileW(W, Ws[0], colBase, K, 0, t);
  wait_async0();
  __syncthreads();

  int cur = 0;
  for (int kb = 0; kb < K; kb += 32) {
    int nxt = cur ^ 1;
    if (kb + 32 < K) {
      stage_tileA(A, As[nxt], rowBase, K, kb + 32, t);
      stage_tileW(W, Ws[nxt], colBase, K, kb + 32, t);
    }

    Frag a[2], b[2];
    #pragma unroll
    for (int i = 0; i < 2; ++i) {
      int row = wm * 32 + i * 16 + l16;
      a[i].q[0] = *(const uint4*)(As[cur] + row * LDT + half * 8);
      a[i].q[1] = *(const uint4*)(As[cur] + row * LDT + 16 + half * 8);
    }
    #pragma unroll
    for (int j = 0; j < 2; ++j) {
      int n = wn * 32 + j * 16 + l16;
      b[j].q[0] = *(const uint4*)(Ws[cur] + n * LDT + half * 8);
      b[j].q[1] = *(const uint4*)(Ws[cur] + n * LDT + 16 + half * 8);
    }
    #pragma unroll
    for (int i = 0; i < 2; ++i)
      #pragma unroll
      for (int j = 0; j < 2; ++j)
        acc[i][j] = wmma_f16(a[i], b[j], acc[i][j]);

    wait_async0();
    __syncthreads();
    cur = nxt;
  }

  #pragma unroll
  for (int i = 0; i < 2; ++i) {
    #pragma unroll
    for (int r = 0; r < 8; ++r) {
      long m = rowBase + wm * 32 + i * 16 + r + half * 8;   // edge id
      long so = (long)src[m] * C_;
      long dofs = (long)dst[m] * C_;
      #pragma unroll
      for (int j = 0; j < 2; ++j) {
        int n = (int)colBase + wn * 32 + j * 16 + l16;
        float v = acc[i][j][r] + bias[n] + h[so + n];
        v = fmaxf(v, 0.0f);
        atomic_add_f32(&agg[dofs + n], v);
      }
    }
  }
}

// ---------------------------------------------------------------------------
// Fused attention: one workgroup per (graph, head).
// S = QK^T/8 (WMMA) -> softmax (LDS) -> O = P V (WMMA)
// ---------------------------------------------------------------------------
__global__ __launch_bounds__(256) void attn_kernel(
    const _Float16* __restrict__ qkv, _Float16* __restrict__ attn)
{
  extern __shared__ char smem[];
  _Float16* qs  = (_Float16*)smem;                  // [128][72]
  _Float16* ks  = qs + 128 * 72;                    // [128][72]
  _Float16* vsT = ks + 128 * 72;                    // [64][136]  (transposed V)
  float*    S   = (float*)(vsT + 64 * 136);         // [128][132]
  _Float16* Ps  = (_Float16*)(S + 128 * 132);       // [128][136]

  const int g  = blockIdx.x >> 2;
  const int hh = blockIdx.x & 3;
  const int t  = threadIdx.x;

  // stage Q, K, V^T
  for (int i = t; i < 128 * 64; i += 256) {
    int nrow = i >> 6, d = i & 63;
    long base = ((long)(g * N_ + nrow)) * (3 * C_);
    qs[nrow * 72 + d]   = qkv[base +            hh * DH_ + d];
    ks[nrow * 72 + d]   = qkv[base + C_ +       hh * DH_ + d];
    vsT[d * 136 + nrow] = qkv[base + 2 * C_ +   hh * DH_ + d];
  }
  __syncthreads();

  const int lane = t & 31, w = t >> 5, half = lane >> 4, l16 = lane & 15;

  // S = Q K^T * (1/sqrt(64)); wave w owns rows [w*16, w*16+16)
  {
    int row = w * 16 + l16;
    Frag a[2];
    #pragma unroll
    for (int kk2 = 0; kk2 < 2; ++kk2) {
      a[kk2].q[0] = *(const uint4*)(qs + row * 72 + kk2 * 32 + half * 8);
      a[kk2].q[1] = *(const uint4*)(qs + row * 72 + kk2 * 32 + 16 + half * 8);
    }
    for (int tile = 0; tile < 8; ++tile) {
      v8f s = {};
      int n = tile * 16 + l16;
      #pragma unroll
      for (int kk2 = 0; kk2 < 2; ++kk2) {
        Frag b;
        b.q[0] = *(const uint4*)(ks + n * 72 + kk2 * 32 + half * 8);
        b.q[1] = *(const uint4*)(ks + n * 72 + kk2 * 32 + 16 + half * 8);
        s = wmma_f16(a[kk2], b, s);
      }
      #pragma unroll
      for (int r = 0; r < 8; ++r)
        S[(w * 16 + r + half * 8) * 132 + tile * 16 + l16] = s[r] * 0.125f;
    }
  }
  __syncthreads();

  // row softmax; thread r owns row r
  if (t < 128) {
    float mx = -1e30f;
    for (int c = 0; c < 128; ++c) mx = fmaxf(mx, S[t * 132 + c]);
    float sum = 0.0f;
    for (int c = 0; c < 128; ++c) {
      float e = __expf(S[t * 132 + c] - mx);
      S[t * 132 + c] = e;
      sum += e;
    }
    float inv = 1.0f / sum;
    for (int c = 0; c < 128; ++c)
      Ps[t * 136 + c] = (_Float16)(S[t * 132 + c] * inv);
  }
  __syncthreads();

  // O = P V; wave w owns rows [w*16, w*16+16), cols 0..63
  {
    int row = w * 16 + l16;
    for (int tile = 0; tile < 4; ++tile) {
      v8f o = {};
      int n = tile * 16 + l16;
      #pragma unroll
      for (int kk = 0; kk < 128; kk += 32) {
        Frag a, b;
        a.q[0] = *(const uint4*)(Ps + row * 136 + kk + half * 8);
        a.q[1] = *(const uint4*)(Ps + row * 136 + kk + 16 + half * 8);
        b.q[0] = *(const uint4*)(vsT + n * 136 + kk + half * 8);
        b.q[1] = *(const uint4*)(vsT + n * 136 + kk + 16 + half * 8);
        o = wmma_f16(a, b, o);
      }
      #pragma unroll
      for (int r = 0; r < 8; ++r) {
        int nrow = w * 16 + r + half * 8;
        long node = (long)g * N_ + nrow;
        attn[node * C_ + hh * DH_ + tile * 16 + l16] = (_Float16)o[r];
      }
    }
  }
}

// ---------------------------------------------------------------------------
// Elementwise / reduction helper kernels
// ---------------------------------------------------------------------------
__global__ void embed_nodes(const int* __restrict__ x,
    const float* e0, const float* e1, const float* e2, const float* e3,
    const float* e4, const float* e5, const float* e6, const float* e7,
    const float* e8, float* __restrict__ h, _Float16* __restrict__ hH)
{
  int node = blockIdx.x, c = threadIdx.x;
  const int* xr = x + node * 9;
  float v = e0[xr[0] * C_ + c] + e1[xr[1] * C_ + c] + e2[xr[2] * C_ + c]
          + e3[xr[3] * C_ + c] + e4[xr[4] * C_ + c] + e5[xr[5] * C_ + c]
          + e6[xr[6] * C_ + c] + e7[xr[7] * C_ + c] + e8[xr[8] * C_ + c];
  long i = (long)node * C_ + c;
  h[i] = v;
  hH[i] = (_Float16)v;
}

__global__ void embed_edges(const int* __restrict__ ea,
    const float* b0, const float* b1, const float* b2,
    _Float16* __restrict__ eH)
{
  int ed = blockIdx.x, c = threadIdx.x;
  const int* er = ea + ed * 3;
  float v = b0[er[0] * C_ + c] + b1[er[1] * C_ + c] + b2[er[2] * C_ + c];
  eH[(long)ed * C_ + c] = (_Float16)v;
}

__global__ void make_zc0(const float* __restrict__ h, const float* __restrict__ agg,
                         const float* __restrict__ epsArr, int l,
                         _Float16* __restrict__ out)
{
  long i = (long)blockIdx.x * 256 + threadIdx.x;
  float eps1 = 1.0f + epsArr[l];
  out[i] = (_Float16)(eps1 * h[i] + agg[i]);
}

__global__ void add2(const float* __restrict__ a, const float* __restrict__ b,
                     float* __restrict__ outF, _Float16* __restrict__ outH)
{
  long i = (long)blockIdx.x * 256 + threadIdx.x;
  float v = a[i] + b[i];
  outF[i] = v;
  outH[i] = (_Float16)v;
}

__global__ __launch_bounds__(256) void bn_stats(
    const float* __restrict__ X, const float* __restrict__ res,
    int M, int Cn, float* __restrict__ mean, float* __restrict__ rstd)
{
  int c = blockIdx.x;
  float s = 0.0f, s2 = 0.0f;
  for (int r = threadIdx.x; r < M; r += 256) {
    float v = X[(long)r * Cn + c];
    if (res) v += res[(long)r * Cn + c];
    s += v; s2 += v * v;
  }
  __shared__ float sh[256], sh2[256];
  sh[threadIdx.x] = s; sh2[threadIdx.x] = s2;
  __syncthreads();
  for (int o = 128; o > 0; o >>= 1) {
    if (threadIdx.x < o) {
      sh[threadIdx.x]  += sh[threadIdx.x + o];
      sh2[threadIdx.x] += sh2[threadIdx.x + o];
    }
    __syncthreads();
  }
  if (threadIdx.x == 0) {
    float m = sh[0] / (float)M;
    float var = sh2[0] / (float)M - m * m;
    mean[c] = m;
    rstd[c] = rsqrtf(var + 1e-5f);
  }
}

__global__ void bn_apply(const float* __restrict__ X, const float* __restrict__ res,
                         const float* __restrict__ mean, const float* __restrict__ rstd,
                         const float* __restrict__ gam, const float* __restrict__ bet,
                         float* __restrict__ outF, _Float16* __restrict__ outH,
                         int Cn, int relu)
{
  long i = (long)blockIdx.x * 256 + threadIdx.x;
  int c = (int)(i % Cn);
  float v = X[i];
  if (res) v += res[i];
  v = (v - mean[c]) * rstd[c] * gam[c] + bet[c];
  if (relu) v = fmaxf(v, 0.0f);
  if (outF) outF[i] = v;
  if (outH) outH[i] = (_Float16)v;
}

__global__ void pool_kernel(const float* __restrict__ h, _Float16* __restrict__ out)
{
  int g = blockIdx.x, c = threadIdx.x;
  float s = 0.0f;
  for (int n = 0; n < N_; ++n) s += h[((long)g * N_ + n) * C_ + c];
  out[(long)g * C_ + c] = (_Float16)s;
}

__global__ __launch_bounds__(256) void l2norm(const float* __restrict__ x,
                                              float* __restrict__ out)
{
  int g = blockIdx.x;
  __shared__ float sh[256];
  float s = 0.0f;
  for (int c = threadIdx.x; c < OUT_; c += 256) {
    float v = x[(long)g * OUT_ + c];
    s += v * v;
  }
  sh[threadIdx.x] = s;
  __syncthreads();
  for (int o = 128; o > 0; o >>= 1) {
    if (threadIdx.x < o) sh[threadIdx.x] += sh[threadIdx.x + o];
    __syncthreads();
  }
  float inv = 1.0f / fmaxf(sqrtf(sh[0]), 1e-12f);
  for (int c = threadIdx.x; c < OUT_; c += 256)
    out[(long)g * OUT_ + c] = x[(long)g * OUT_ + c] * inv;
}

__global__ void f2h(const float* __restrict__ s, _Float16* __restrict__ d, long n)
{
  long i = (long)blockIdx.x * 256 + threadIdx.x;
  if (i < n) d[i] = (_Float16)s[i];
}

// ---------------------------------------------------------------------------
// host orchestration
// ---------------------------------------------------------------------------
extern "C" void kernel_launch(void* const* d_in, const int* in_sizes, int n_in,
                              void* d_out, int out_size, void* d_ws, size_t ws_size,
                              hipStream_t stream) {
  (void)in_sizes; (void)n_in; (void)out_size; (void)ws_size;

  // ----- inputs (setup_inputs dict order) -----
  const float* atom_emb[9];
  for (int i = 0; i < 9; ++i) atom_emb[i] = (const float*)d_in[i];
  const float* bond_emb[3];
  for (int i = 0; i < 3; ++i) bond_emb[i] = (const float*)d_in[9 + i];
  const float* epsArr     = (const float*)d_in[12];
  const float* gine_w     = (const float*)d_in[13];
  const float* gine_b     = (const float*)d_in[14];
  const float* nn_w1      = (const float*)d_in[15];
  const float* nn_b1      = (const float*)d_in[16];
  const float* nn_bn_g    = (const float*)d_in[17];
  const float* nn_bn_b    = (const float*)d_in[18];
  const float* nn_w2      = (const float*)d_in[19];
  const float* nn_b2      = (const float*)d_in[20];
  const float* attn_in_w  = (const float*)d_in[21];
  const float* attn_in_b  = (const float*)d_in[22];
  const float* attn_out_w = (const float*)d_in[23];
  const float* attn_out_b = (const float*)d_in[24];
  const float* mlp_w1     = (const float*)d_in[25];
  const float* mlp_b1     = (const float*)d_in[26];
  const float* mlp_w2     = (const float*)d_in[27];
  const float* mlp_b2     = (const float*)d_in[28];
  const float* bn1_g      = (const float*)d_in[29];
  const float* bn1_b      = (const float*)d_in[30];
  const float* bn2_g      = (const float*)d_in[31];
  const float* bn2_b      = (const float*)d_in[32];
  const float* bn3_g      = (const float*)d_in[33];
  const float* bn3_b      = (const float*)d_in[34];
  const float* proj_w1    = (const float*)d_in[35];
  const float* proj_b1    = (const float*)d_in[36];
  const float* proj_w2    = (const float*)d_in[37];
  const float* proj_b2    = (const float*)d_in[38];
  const int*   x          = (const int*)d_in[39];
  const int*   edge_attr  = (const int*)d_in[40];
  const int*   eidx       = (const int*)d_in[41];
  const int*   src = eidx;
  const int*   dst = eidx + E_;
  float* outp = (float*)d_out;

  // ----- workspace carve-up -----
  char* p = (char*)d_ws;
  auto alloc = [&](size_t bytes) -> char* {
    char* r = p;
    p += (bytes + 255) & ~(size_t)255;
    return r;
  };
  float*    h      = (float*)   alloc((size_t)GN_ * C_ * 4);
  _Float16* hH     = (_Float16*)alloc((size_t)GN_ * C_ * 2);
  _Float16* eH     = (_Float16*)alloc((size_t)E_  * C_ * 2);
  float*    agg    = (float*)   alloc((size_t)GN_ * C_ * 4);   // also reused as h1+h2 sum
  float*    tmpF   = (float*)   alloc((size_t)GN_ * C_ * 4);
  _Float16* tmpH   = (_Float16*)alloc((size_t)GN_ * C_ * 2);
  float*    h1     = (float*)   alloc((size_t)GN_ * C_ * 4);
  float*    h2     = (float*)   alloc((size_t)GN_ * C_ * 4);
  float*    wide   = (float*)   alloc((size_t)GN_ * 2 * C_ * 4);
  _Float16* wideH  = (_Float16*)alloc((size_t)GN_ * 2 * C_ * 2);
  _Float16* qkvH   = (_Float16*)alloc((size_t)GN_ * 3 * C_ * 2);
  _Float16* attnH  = (_Float16*)alloc((size_t)GN_ * C_ * 2);
  float*    meanB  = (float*)   alloc(512 * 4);
  float*    rstdB  = (float*)   alloc(512 * 4);
  _Float16* poolH  = (_Float16*)alloc((size_t)G_ * C_ * 2);
  _Float16* pool2H = (_Float16*)alloc((size_t)G_ * C_ * 2);
  float*    logits = (float*)   alloc((size_t)G_ * OUT_ * 4);
  // f16 weights
  _Float16* gineWH   = (_Float16*)alloc((size_t)L_ * C_ * C_ * 2);
  _Float16* nnW1H    = (_Float16*)alloc((size_t)L_ * 2 * C_ * C_ * 2);
  _Float16* nnW2H    = (_Float16*)alloc((size_t)L_ * C_ * 2 * C_ * 2);
  _Float16* attnInWH = (_Float16*)alloc((size_t)L_ * 3 * C_ * C_ * 2);
  _Float16* attnOutWH= (_Float16*)alloc((size_t)L_ * C_ * C_ * 2);
  _Float16* mlpW1H   = (_Float16*)alloc((size_t)L_ * 2 * C_ * C_ * 2);
  _Float16* mlpW2H   = (_Float16*)alloc((size_t)L_ * C_ * 2 * C_ * 2);
  _Float16* projW1H  = (_Float16*)alloc((size_t)C_ * C_ * 2);
  _Float16* projW2H  = (_Float16*)alloc((size_t)OUT_ * C_ * 2);

  auto conv = [&](const float* s, _Float16* d, long n) {
    f2h<<<dim3((unsigned)((n + 255) / 256)), dim3(256), 0, stream>>>(s, d, n);
  };
  conv(gine_w,     gineWH,    (long)L_ * C_ * C_);
  conv(nn_w1,      nnW1H,     (long)L_ * 2 * C_ * C_);
  conv(nn_w2,      nnW2H,     (long)L_ * C_ * 2 * C_);
  conv(attn_in_w,  attnInWH,  (long)L_ * 3 * C_ * C_);
  conv(attn_out_w, attnOutWH, (long)L_ * C_ * C_);
  conv(mlp_w1,     mlpW1H,    (long)L_ * 2 * C_ * C_);
  conv(mlp_w2,     mlpW2H,    (long)L_ * C_ * 2 * C_);
  conv(proj_w1,    projW1H,   (long)C_ * C_);
  conv(proj_w2,    projW2H,   (long)OUT_ * C_);

  // ----- encoders -----
  embed_nodes<<<dim3(GN_), dim3(C_), 0, stream>>>(x,
      atom_emb[0], atom_emb[1], atom_emb[2], atom_emb[3], atom_emb[4],
      atom_emb[5], atom_emb[6], atom_emb[7], atom_emb[8], h, hH);
  embed_edges<<<dim3(E_), dim3(C_), 0, stream>>>(edge_attr,
      bond_emb[0], bond_emb[1], bond_emb[2], eH);

  auto gemm = [&](const _Float16* A, const _Float16* W, const float* bias,
                  float* oF, _Float16* oH, const float* res,
                  int M, int N, int K, int relu) {
    gemm_f16<<<dim3((unsigned)(M / TM), (unsigned)(N / TN)), dim3(256), 0, stream>>>(
        A, W, bias, oF, oH, res, M, N, K, relu);
  };

  const size_t attnSmem = (size_t)128 * 72 * 2 * 2   // qs, ks
                        + (size_t)64 * 136 * 2       // vsT
                        + (size_t)128 * 132 * 4      // S
                        + (size_t)128 * 136 * 2;     // Ps

  const unsigned nodeBlocks  = (unsigned)((long)GN_ * C_ / 256);
  const unsigned wideBlocks  = (unsigned)((long)GN_ * 2 * C_ / 256);

  for (int l = 0; l < L_; ++l) {
    // --- GINEConv: edge GEMM + fused relu/scatter-add ---
    hipMemsetAsync(agg, 0, (size_t)GN_ * C_ * 4, stream);
    gemm_gine<<<dim3(E_ / TM, C_ / TN), dim3(256), 0, stream>>>(
        eH, gineWH + (long)l * C_ * C_, gine_b + (long)l * C_,
        h, src, dst, agg, E_, C_);
    // zc0 = (1+eps)*h + agg   (f16, feeds nn GEMM)
    make_zc0<<<nodeBlocks, 256, 0, stream>>>(h, agg, epsArr, l, tmpH);
    // nn: wide = zc0 @ W1^T + b1 ; BN ; relu
    gemm(tmpH, nnW1H + (long)l * 2 * C_ * C_, nn_b1 + (long)l * 2 * C_,
         wide, nullptr, nullptr, GN_, 2 * C_, C_, 0);
    bn_stats<<<2 * C_, 256, 0, stream>>>(wide, nullptr, GN_, 2 * C_, meanB, rstdB);
    bn_apply<<<wideBlocks, 256, 0, stream>>>(wide, nullptr, meanB, rstdB,
        nn_bn_g + (long)l * 2 * C_, nn_bn_b + (long)l * 2 * C_,
        nullptr, wideH, 2 * C_, 1);
    // zc = wide @ W2^T + b2 ; h1 = BN(zc + h)
    gemm(wideH, nnW2H + (long)l * C_ * 2 * C_, nn_b2 + (long)l * C_,
         tmpF, nullptr, nullptr, GN_, C_, 2 * C_, 0);
    bn_stats<<<C_, 256, 0, stream>>>(tmpF, h, GN_, C_, meanB, rstdB);
    bn_apply<<<nodeBlocks, 256, 0, stream>>>(tmpF, h, meanB, rstdB,
        bn1_g + (long)l * C_, bn1_b + (long)l * C_, h1, nullptr, C_, 0);

    // --- attention ---
    gemm(hH, attnInWH + (long)l * 3 * C_ * C_, attn_in_b + (long)l * 3 * C_,
         nullptr, qkvH, nullptr, GN_, 3 * C_, C_, 0);
    attn_kernel<<<dim3(G_ * H_), dim3(256), attnSmem, stream>>>(qkvH, attnH);
    gemm(attnH, attnOutWH + (long)l * C_ * C_, attn_out_b + (long)l * C_,
         tmpF, nullptr, nullptr, GN_, C_, C_, 0);
    bn_stats<<<C_, 256, 0, stream>>>(tmpF, h, GN_, C_, meanB, rstdB);
    bn_apply<<<nodeBlocks, 256, 0, stream>>>(tmpF, h, meanB, rstdB,
        bn2_g + (long)l * C_, bn2_b + (long)l * C_, h2, nullptr, C_, 0);

    // --- combine + FFN ---
    add2<<<nodeBlocks, 256, 0, stream>>>(h1, h2, agg, tmpH);  // agg := h1+h2 (f32), tmpH f16
    gemm(tmpH, mlpW1H + (long)l * 2 * C_ * C_, mlp_b1 + (long)l * 2 * C_,
         nullptr, wideH, nullptr, GN_, 2 * C_, C_, 1);        // relu in epilogue
    gemm(wideH, mlpW2H + (long)l * C_ * 2 * C_, mlp_b2 + (long)l * C_,
         tmpF, nullptr, nullptr, GN_, C_, 2 * C_, 0);
    bn_stats<<<C_, 256, 0, stream>>>(tmpF, agg, GN_, C_, meanB, rstdB);
    bn_apply<<<nodeBlocks, 256, 0, stream>>>(tmpF, agg, meanB, rstdB,
        bn3_g + (long)l * C_, bn3_b + (long)l * C_, h, hH, C_, 0);
  }

  // ----- pool + projection head + L2 normalize -----
  pool_kernel<<<dim3(G_), dim3(C_), 0, stream>>>(h, poolH);
  gemm(poolH, projW1H, proj_b1, nullptr, pool2H, nullptr, G_, C_, C_, 1);
  gemm(pool2H, projW2H, proj_b2, logits, nullptr, nullptr, G_, OUT_, C_, 0);
  l2norm<<<dim3(G_), dim3(256), 0, stream>>>(logits, outp);
}
